// highwayNet_34256659152924
// MI455X (gfx1250) — matrix operand order repaired and checked by reading, not compiled, not run on hardware
//
#include <hip/hip_runtime.h>
#include <hip/hip_bf16.h>
#include <math.h>

// ---------------------------------------------------------------------------
// highwayNet-style encoder/attention/decoder for MI455X (gfx1250, wave32).
// All large GEMMs (LSTM gates, q/k/v, decoder) run on v_wmma_f32_16x16x32_f16
// with f32 accumulation; each wave computes a 16x64 strip (A-fragment reused
// across 4 WMMAs per K-step) to raise FLOP/byte out of L2.
// ---------------------------------------------------------------------------

typedef __attribute__((ext_vector_type(16))) _Float16 v16h;
typedef __attribute__((ext_vector_type(8)))  _Float16 v8h;
typedef __attribute__((ext_vector_type(8)))  float    v8f;

constexpr int kENC = 512, kDEC = 512, kEMB = 32, kNH = 8;
constexpr int kT = 16, kOUT = 25, kB = 128, kGH = 3, kGW = 13, kNNB = 1664;
constexpr int kG4 = 4 * kENC;   // 2048 gate width

__device__ __forceinline__ float sigmoidf_(float x) { return 1.0f / (1.0f + __expf(-x)); }
__device__ __forceinline__ float lrelu_(float x)    { return x > 0.0f ? x : 0.1f * x; }

// ------------------------- generic WMMA GEMM -------------------------------
// C[m][n] = sum_k A[m][k] * W[n][k]  (+ bias[n]) (+ addend[m][n])
// A: (M,K) f16 row-major, W: (N,K) f16 row-major, C/addend: (M,N) f32.
// One wave per 16x64 strip (4 accumulators); M mult of 16, N mult of 64,
// K mult of 32. Tile exit is wave-uniform so EXEC is all-1s at every WMMA.
__global__ __launch_bounds__(256) void gemm_wmma_f16(
    const _Float16* __restrict__ A, const _Float16* __restrict__ W,
    float* __restrict__ C, const float* __restrict__ addend,
    const float* __restrict__ bias, int M, int N, int K)
{
  const int lane = threadIdx.x & 31;
  const int wave = threadIdx.x >> 5;
  const int tilesN = N >> 6;                       // strips of 64 columns
  const int tile = blockIdx.x * 8 + wave;
  if (tile >= (M >> 4) * tilesN) return;           // wave-uniform exit
  const int m0 = (tile / tilesN) << 4;
  const int n0 = (tile % tilesN) << 6;
  const int half = lane >> 4, r = lane & 15;

  const _Float16* arow = A + (size_t)(m0 + r) * K;
  const _Float16* wrow0 = W + (size_t)(n0 + r) * K;
  const _Float16* wrow1 = wrow0 + (size_t)16 * K;
  const _Float16* wrow2 = wrow0 + (size_t)32 * K;
  const _Float16* wrow3 = wrow0 + (size_t)48 * K;

  v8f acc0 = {}, acc1 = {}, acc2 = {}, acc3 = {};
  for (int kb = 0; kb < K; kb += 32) {
    // A 16x32 per-lane layout: half 0 -> K {0..7,16..23}; half 1 -> K {8..15,24..31}
    v8h a0 = *(const v8h*)(arow + kb + half * 8);
    v8h a1 = *(const v8h*)(arow + kb + 16 + half * 8);
    v16h av;
#pragma unroll
    for (int i = 0; i < 8; ++i) { av[i] = a0[i]; av[i + 8] = a1[i]; }
    // B 32x16 per-lane layout: column n + r; half 0 -> K {0..7,16..23}; half 1 -> K {8..15,24..31}
    const int bo = kb + half * 8;
#pragma unroll
    for (int nn = 0; nn < 4; ++nn) {
      const _Float16* wr = (nn == 0) ? wrow0 : (nn == 1) ? wrow1 : (nn == 2) ? wrow2 : wrow3;
      v8h b0 = *(const v8h*)(wr + bo);
      v8h b1 = *(const v8h*)(wr + bo + 16);
      v16h bv;
#pragma unroll
      for (int i = 0; i < 8; ++i) { bv[i] = b0[i]; bv[i + 8] = b1[i]; }
      v8f& acc = (nn == 0) ? acc0 : (nn == 1) ? acc1 : (nn == 2) ? acc2 : acc3;
      acc = __builtin_amdgcn_wmma_f32_16x16x32_f16(false, av, false, bv,
                                                   (short)0, acc, false, false);
    }
  }
  // D layout: VGPR j -> (M = m0 + j + 8*half, N = n + r)
#pragma unroll
  for (int nn = 0; nn < 4; ++nn) {
    const v8f& acc = (nn == 0) ? acc0 : (nn == 1) ? acc1 : (nn == 2) ? acc2 : acc3;
    const int n = n0 + nn * 16 + r;
#pragma unroll
    for (int j = 0; j < 8; ++j) {
      const int m = m0 + j + 8 * half;
      float v = acc[j];
      if (bias)   v += bias[n];
      if (addend) v += addend[(size_t)m * N + n];
      C[(size_t)m * N + n] = v;
    }
  }
}

// ------------------------- elementwise kernels -----------------------------
__global__ void f32_to_f16_k(const float* __restrict__ s, _Float16* __restrict__ d, int n) {
  int i = blockIdx.x * blockDim.x + threadIdx.x;
  if (i < n) d[i] = (_Float16)s[i];
}
__global__ void sum2_k(const float* a, const float* b, float* o, int n) {
  int i = blockIdx.x * blockDim.x + threadIdx.x;
  if (i < n) o[i] = a[i] + b[i];
}
__global__ void zero_f32_k(float* p, int n) {
  int i = blockIdx.x * blockDim.x + threadIdx.x;
  if (i < n) p[i] = 0.0f;
}
__global__ void zero_f16_k(_Float16* p, int n) {
  int i = blockIdx.x * blockDim.x + threadIdx.x;
  if (i < n) p[i] = (_Float16)0.0f;
}

// res = g_seq + hist_tem   -> f16 rows (t*B+b, 32)
__global__ void prep_res_k(const float* __restrict__ hist, const float* __restrict__ graph,
                           const float* __restrict__ pos, const float* __restrict__ Wg1,
                           const float* __restrict__ bg1, const float* __restrict__ Wg2,
                           const float* __restrict__ bg2, const float* __restrict__ Wip,
                           const float* __restrict__ bip, _Float16* __restrict__ X) {
  int id = blockIdx.x * blockDim.x + threadIdx.x;
  if (id >= kT * kB) return;
  int t = id / kB, b = id % kB;
  float s = bg1[t];
  for (int j = 0; j < kGH * kGW; ++j) {               // j = gw*GH + gh (transpose 0,2,1)
    int gw = j / kGH, gh = j % kGH;
    float gv = graph[(b * kGH + gh) * kGW + gw] + pos[(b * kGH + gh) * kGW + gw];
    s += gv * Wg1[t * (kGH * kGW) + j];
  }
  s = lrelu_(s);
  float h0 = hist[(size_t)id * 2], h1 = hist[(size_t)id * 2 + 1];
  for (int e = 0; e < kEMB; ++e) {
    float gse = lrelu_(s * Wg2[e] + bg2[e]);
    float ht  = lrelu_(h0 * Wip[e * 2] + h1 * Wip[e * 2 + 1] + bip[e]);
    X[(size_t)id * kEMB + e] = (_Float16)(gse + ht);
  }
}

__global__ void prep_nbr_k(const float* __restrict__ nbrs, const float* __restrict__ Wip,
                           const float* __restrict__ bip, _Float16* __restrict__ X) {
  int id = blockIdx.x * blockDim.x + threadIdx.x;
  if (id >= kT * kNNB) return;
  float n0 = nbrs[(size_t)id * 2], n1 = nbrs[(size_t)id * 2 + 1];
  for (int e = 0; e < kEMB; ++e)
    X[(size_t)id * kEMB + e] = (_Float16)lrelu_(n0 * Wip[e * 2] + n1 * Wip[e * 2 + 1] + bip[e]);
}

// gates (Mb, 2048) in i,f,g,o order -> update (h,c); h also to f16 and optional seq f16
__global__ void lstm_cell_k(const float* __restrict__ g, float* __restrict__ h,
                            float* __restrict__ c, _Float16* __restrict__ hH,
                            _Float16* __restrict__ seq, long seqRowStride, int Mb) {
  int i = blockIdx.x * blockDim.x + threadIdx.x;
  if (i >= Mb * kENC) return;
  int m = i / kENC, e = i % kENC;
  const float* gr = g + (size_t)m * kG4;
  float gi = gr[e], gf = gr[kENC + e], gg = gr[2 * kENC + e], go = gr[3 * kENC + e];
  float cs = sigmoidf_(gf) * c[i] + sigmoidf_(gi) * tanhf(gg);
  float hs = sigmoidf_(go) * tanhf(cs);
  c[i] = cs; h[i] = hs; hH[i] = (_Float16)hs;
  if (seq) seq[(size_t)m * seqRowStride + e] = (_Float16)hs;
}

// small multi-head attention over time: one thread per (b, head, row i)
__global__ void attn_time_k(const float* __restrict__ q, const float* __restrict__ k,
                            const float* __restrict__ v, float* __restrict__ att) {
  int id = blockIdx.x * blockDim.x + threadIdx.x;
  if (id >= kB * kNH * kT) return;
  int i = id & 15, bh = id >> 4, h = bh & (kNH - 1), b = bh >> 3;
  const int dk = kENC / kNH;  // 64
  const float* qr = q + ((size_t)(b * kT + i) * kENC) + h * dk;
  float sc[kT], mx = -1e30f;
#pragma unroll
  for (int j = 0; j < kT; ++j) {
    const float* kr = k + ((size_t)(b * kT + j) * kENC) + h * dk;
    float d = 0.0f;
    for (int dd = 0; dd < dk; ++dd) d += qr[dd] * kr[dd];
    sc[j] = d * 0.125f;                // 1/sqrt(64)
    mx = fmaxf(mx, sc[j]);
  }
  float sum = 0.0f;
#pragma unroll
  for (int j = 0; j < kT; ++j) { sc[j] = __expf(sc[j] - mx); sum += sc[j]; }
  float inv = 1.0f / sum;
  float* ar = att + (size_t)(b * kT + i) * kENC + h * dk;
  for (int dd = 0; dd < dk; ++dd) {
    float a = 0.0f;
#pragma unroll
    for (int j = 0; j < kT; ++j) a += sc[j] * v[(size_t)(b * kT + j) * kENC + h * dk + dd];
    ar[dd] = a * inv;
  }
}

__global__ void new_hidden_k(const float* __restrict__ att, const float* __restrict__ Wpre2,
                             const float* __restrict__ bpre2, float* __restrict__ nh) {
  int id = blockIdx.x * blockDim.x + threadIdx.x;
  if (id >= kB * kENC) return;
  int b = id / kENC, e = id % kENC;
  float s = bpre2[0];
#pragma unroll
  for (int t = 0; t < kT; ++t) s += att[(size_t)(b * kT + t) * kENC + e] * Wpre2[t];
  nh[id] = s;
}

__global__ void nbr_w_k(const _Float16* __restrict__ noH, const float* __restrict__ Wpre4,
                        const float* __restrict__ bpre4, float* __restrict__ w) {
  int id = blockIdx.x * blockDim.x + threadIdx.x;  // n*T + t
  if (id >= kNNB * kT) return;
  const _Float16* row = noH + (size_t)id * kENC;
  float s = bpre4[0];
  for (int e = 0; e < kENC; ++e) s += tanhf((float)row[e]) * Wpre4[e];
  w[id] = s;
}

__global__ void softmax_rows_k(const float* __restrict__ src, float* __restrict__ dA,
                               float* __restrict__ dB, int rows, int width) {
  int r = blockIdx.x * blockDim.x + threadIdx.x;
  if (r >= rows) return;
  const float* s = src + (size_t)r * width;
  float mx = -1e30f;
  for (int j = 0; j < width; ++j) mx = fmaxf(mx, s[j]);
  float sum = 0.0f;
  for (int j = 0; j < width; ++j) sum += __expf(s[j] - mx);
  float inv = 1.0f / sum;
  for (int j = 0; j < width; ++j) {
    float a = __expf(s[j] - mx) * inv;
    dA[(size_t)r * width + j] = a;
    dB[(size_t)r * width + j] = a;
  }
}

__global__ void pool_nbr_k(const _Float16* __restrict__ noH, const float* __restrict__ alpha,
                           float* __restrict__ enc) {
  int id = blockIdx.x * blockDim.x + threadIdx.x;
  if (id >= kNNB * kENC) return;
  int n = id / kENC, e = id % kENC;
  float s = 0.0f;
#pragma unroll
  for (int t = 0; t < kT; ++t) s += (float)noH[(size_t)(n * kT + t) * kENC + e] * alpha[n * kT + t];
  enc[id] = fmaxf(s, 0.0f);
}

// serial prefix scan over 4992 occupancy cells (masked_scatter ordering)
__global__ void scan_mask_k(const unsigned char* __restrict__ masks, int* __restrict__ idx) {
  if (threadIdx.x != 0 || blockIdx.x != 0) return;
  int cnt = 0;
  for (int c = 0; c < kB * kGH * kGW; ++c) {
    idx[c] = masks[(size_t)c * kENC] ? cnt++ : -1;
  }
}

__global__ void scatter_soc_k(const int* __restrict__ idx, const float* __restrict__ enc,
                              float* __restrict__ new_hs) {
  int id = blockIdx.x * blockDim.x + threadIdx.x;   // cell*ENC + e
  if (id >= kB * kGH * kGW * kENC) return;
  int c = id / kENC, e = id % kENC;
  int src = idx[c];
  if (src < 0) return;
  int b = c / (kGH * kGW), rem = c % (kGH * kGW);
  int gh = rem / kGW, gw = rem % kGW;
  int j = gw * kGH + gh;                             // transpose (0,3,2,1) flatten
  new_hs[((size_t)b * kENC + e) * 40 + j] = enc[(size_t)src * kENC + e];
}

__global__ void set_hidden_col_k(const float* __restrict__ nh, float* __restrict__ new_hs) {
  int id = blockIdx.x * blockDim.x + threadIdx.x;
  if (id >= kB * kENC) return;
  int b = id / kENC, e = id % kENC;
  new_hs[((size_t)b * kENC + e) * 40 + 39] = nh[id];
}

__global__ void w2_k(const float* __restrict__ new_hs, const float* __restrict__ Wpre4,
                     const float* __restrict__ bpre4, float* __restrict__ w2) {
  int id = blockIdx.x * blockDim.x + threadIdx.x;   // b*40 + j
  if (id >= kB * 40) return;
  int b = id / 40, j = id % 40;
  float s = bpre4[0];
  for (int e = 0; e < kENC; ++e) s += tanhf(new_hs[((size_t)b * kENC + e) * 40 + j]) * Wpre4[e];
  w2[id] = s;
}

__global__ void pool_ha_k(const float* __restrict__ new_hs, const float* __restrict__ alpha,
                          _Float16* __restrict__ encH) {
  int id = blockIdx.x * blockDim.x + threadIdx.x;
  if (id >= kB * kENC) return;
  int b = id / kENC, e = id % kENC;
  float s = 0.0f;
#pragma unroll
  for (int j = 0; j < 40; ++j) s += new_hs[((size_t)b * kENC + e) * 40 + j] * alpha[b * 40 + j];
  encH[id] = (_Float16)fmaxf(s, 0.0f);
}

__global__ void dec_out_k(const float* __restrict__ h, const float* __restrict__ Wop,
                          const float* __restrict__ bop, float* __restrict__ out) {
  int id = blockIdx.x * blockDim.x + threadIdx.x;   // b*2 + k
  if (id >= kB * 2) return;
  int b = id >> 1, k2 = id & 1;
  float s = bop[k2];
  const float* hr = h + (size_t)b * kDEC;
  const float* wr = Wop + (size_t)k2 * kDEC;
  for (int e = 0; e < kDEC; ++e) s += hr[e] * wr[e];
  out[b * 2 + k2] = s;
}

// ---------------------------------------------------------------------------
static inline void launch_gemm(hipStream_t s, const _Float16* A, const _Float16* W,
                               float* C, const float* addend, const float* bias,
                               int M, int N, int K) {
  int tiles = (M / 16) * (N / 64);
  gemm_wmma_f16<<<(tiles + 7) / 8, 256, 0, s>>>(A, W, C, addend, bias, M, N, K);
}
static inline dim3 g1(int n, int t = 256) { return dim3((n + t - 1) / t); }

extern "C" void kernel_launch(void* const* d_in, const int* in_sizes, int n_in,
                              void* d_out, int out_size, void* d_ws, size_t ws_size,
                              hipStream_t stream) {
  (void)in_sizes; (void)n_in; (void)out_size;
  const float* hist   = (const float*)d_in[0];
  const float* nbrs   = (const float*)d_in[1];
  const unsigned char* masks = (const unsigned char*)d_in[2];   // bool occupancy
  const float* graph  = (const float*)d_in[5];
  const float* pos    = (const float*)d_in[6];
  const float* Wip    = (const float*)d_in[7];
  const float* bip    = (const float*)d_in[8];
  const float* Wg1    = (const float*)d_in[9];
  const float* bg1    = (const float*)d_in[10];
  const float* Wg2    = (const float*)d_in[11];
  const float* bg2    = (const float*)d_in[12];
  const float* Wih1   = (const float*)d_in[13];
  const float* Whh1   = (const float*)d_in[14];
  const float* bih1   = (const float*)d_in[15];
  const float* bhh1   = (const float*)d_in[16];
  const float* Wq     = (const float*)d_in[17];
  const float* Wk     = (const float*)d_in[18];
  const float* Wv     = (const float*)d_in[19];
  const float* Wpre2  = (const float*)d_in[20];
  const float* bpre2  = (const float*)d_in[21];
  const float* Wpre4  = (const float*)d_in[22];
  const float* bpre4  = (const float*)d_in[23];
  const float* Wihd   = (const float*)d_in[24];
  const float* Whhd   = (const float*)d_in[25];
  const float* bihd   = (const float*)d_in[26];
  const float* bhhd   = (const float*)d_in[27];
  const float* Wop    = (const float*)d_in[28];
  const float* bop    = (const float*)d_in[29];
  float* out = (float*)d_out;           // [fut 6400 | soft_nbrs 26624 | soft_ha 5120]

  // --- bump allocator over d_ws (~115 MB required) ---
  size_t off = 0;
  auto alloc = [&](size_t bytes) -> void* {
    void* p = (char*)d_ws + off;
    off += (bytes + 255) & ~(size_t)255;
    return p;
  };
  _Float16* WihH  = (_Float16*)alloc((size_t)kG4 * kEMB * 2);
  _Float16* WhhH  = (_Float16*)alloc((size_t)kG4 * kENC * 2);
  _Float16* WqH   = (_Float16*)alloc((size_t)kENC * kENC * 2);
  _Float16* WkH   = (_Float16*)alloc((size_t)kENC * kENC * 2);
  _Float16* WvH   = (_Float16*)alloc((size_t)kENC * kENC * 2);
  _Float16* WihdH = (_Float16*)alloc((size_t)kG4 * kENC * 2);
  _Float16* WhhdH = (_Float16*)alloc((size_t)kG4 * kDEC * 2);
  _Float16* Xenc  = (_Float16*)alloc((size_t)kT * kB * kEMB * 2);
  _Float16* Xnbr  = (_Float16*)alloc((size_t)kT * kNNB * kEMB * 2);
  _Float16* hEncH = (_Float16*)alloc((size_t)kB * kENC * 2);
  _Float16* hNbrH = (_Float16*)alloc((size_t)kNNB * kENC * 2);
  _Float16* hDecH = (_Float16*)alloc((size_t)kB * kDEC * 2);
  _Float16* loH   = (_Float16*)alloc((size_t)kB * kT * kENC * 2);
  _Float16* noH   = (_Float16*)alloc((size_t)kNNB * kT * kENC * 2);
  _Float16* encH  = (_Float16*)alloc((size_t)kB * kENC * 2);
  float* bsum1  = (float*)alloc(kG4 * 4);
  float* bsumd  = (float*)alloc(kG4 * 4);
  float* XWenc  = (float*)alloc((size_t)kT * kB * kG4 * 4);
  float* gates  = (float*)alloc((size_t)kNNB * kG4 * 4);
  float* hEnc   = (float*)alloc((size_t)kB * kENC * 4);
  float* cEnc   = (float*)alloc((size_t)kB * kENC * 4);
  float* hNbr   = (float*)alloc((size_t)kNNB * kENC * 4);
  float* cNbr   = (float*)alloc((size_t)kNNB * kENC * 4);
  float* hDec   = (float*)alloc((size_t)kB * kDEC * 4);
  float* cDec   = (float*)alloc((size_t)kB * kDEC * 4);
  float* qbuf   = (float*)alloc((size_t)kB * kT * kENC * 4);
  float* kbuf   = (float*)alloc((size_t)kB * kT * kENC * 4);
  float* vbuf   = (float*)alloc((size_t)kB * kT * kENC * 4);
  float* att    = (float*)alloc((size_t)kB * kT * kENC * 4);
  float* nhid   = (float*)alloc((size_t)kB * kENC * 4);
  float* nbrEnc = (float*)alloc((size_t)kNNB * kENC * 4);
  float* wbuf   = (float*)alloc((size_t)kNNB * kT * 4);
  float* aNbr   = (float*)alloc((size_t)kNNB * kT * 4);
  float* newHs  = (float*)alloc((size_t)kB * kENC * 40 * 4);
  float* w2buf  = (float*)alloc((size_t)kB * 40 * 4);
  float* aHa    = (float*)alloc((size_t)kB * 40 * 4);
  float* xwd    = (float*)alloc((size_t)kB * kG4 * 4);
  int*   idx    = (int*)alloc((size_t)kB * kGH * kGW * 4);
  (void)ws_size;

  // --- weight conversion + bias sums ---
  f32_to_f16_k<<<g1(kG4 * kEMB), 256, 0, stream>>>(Wih1, WihH, kG4 * kEMB);
  f32_to_f16_k<<<g1(kG4 * kENC), 256, 0, stream>>>(Whh1, WhhH, kG4 * kENC);
  f32_to_f16_k<<<g1(kENC * kENC), 256, 0, stream>>>(Wq, WqH, kENC * kENC);
  f32_to_f16_k<<<g1(kENC * kENC), 256, 0, stream>>>(Wk, WkH, kENC * kENC);
  f32_to_f16_k<<<g1(kENC * kENC), 256, 0, stream>>>(Wv, WvH, kENC * kENC);
  f32_to_f16_k<<<g1(kG4 * kENC), 256, 0, stream>>>(Wihd, WihdH, kG4 * kENC);
  f32_to_f16_k<<<g1(kG4 * kDEC), 256, 0, stream>>>(Whhd, WhhdH, kG4 * kDEC);
  sum2_k<<<g1(kG4), 256, 0, stream>>>(bih1, bhh1, bsum1, kG4);
  sum2_k<<<g1(kG4), 256, 0, stream>>>(bihd, bhhd, bsumd, kG4);

  // --- input embeddings ---
  prep_res_k<<<g1(kT * kB), 256, 0, stream>>>(hist, graph, pos, Wg1, bg1, Wg2, bg2,
                                              Wip, bip, Xenc);
  prep_nbr_k<<<g1(kT * kNNB), 256, 0, stream>>>(nbrs, Wip, bip, Xnbr);

  // --- zero LSTM states ---
  zero_f32_k<<<g1(kB * kENC), 256, 0, stream>>>(hEnc, kB * kENC);
  zero_f32_k<<<g1(kB * kENC), 256, 0, stream>>>(cEnc, kB * kENC);
  zero_f16_k<<<g1(kB * kENC), 256, 0, stream>>>(hEncH, kB * kENC);
  zero_f32_k<<<g1(kNNB * kENC), 256, 0, stream>>>(hNbr, kNNB * kENC);
  zero_f32_k<<<g1(kNNB * kENC), 256, 0, stream>>>(cNbr, kNNB * kENC);
  zero_f16_k<<<g1(kNNB * kENC), 256, 0, stream>>>(hNbrH, kNNB * kENC);
  zero_f32_k<<<g1(kB * kDEC), 256, 0, stream>>>(hDec, kB * kDEC);
  zero_f32_k<<<g1(kB * kDEC), 256, 0, stream>>>(cDec, kB * kDEC);
  zero_f16_k<<<g1(kB * kDEC), 256, 0, stream>>>(hDecH, kB * kDEC);
  zero_f32_k<<<g1(kB * kENC * 40), 256, 0, stream>>>(newHs, kB * kENC * 40);

  // --- encoder LSTM: precompute input projection for all t, then recur ---
  launch_gemm(stream, Xenc, WihH, XWenc, nullptr, bsum1, kT * kB, kG4, kEMB);
  for (int t = 0; t < kT; ++t) {
    launch_gemm(stream, hEncH, WhhH, gates, XWenc + (size_t)t * kB * kG4, nullptr,
                kB, kG4, kENC);
    lstm_cell_k<<<g1(kB * kENC), 256, 0, stream>>>(gates, hEnc, cEnc, hEncH,
                                                   loH + (size_t)t * kENC,
                                                   (long)kT * kENC, kB);
  }

  // --- time attention (q/k/v WMMA GEMMs + scalar softmax attention) ---
  launch_gemm(stream, loH, WqH, qbuf, nullptr, nullptr, kB * kT, kENC, kENC);
  launch_gemm(stream, loH, WkH, kbuf, nullptr, nullptr, kB * kT, kENC, kENC);
  launch_gemm(stream, loH, WvH, vbuf, nullptr, nullptr, kB * kT, kENC, kENC);
  attn_time_k<<<g1(kB * kNH * kT), 256, 0, stream>>>(qbuf, kbuf, vbuf, att);
  new_hidden_k<<<g1(kB * kENC), 256, 0, stream>>>(att, Wpre2, bpre2, nhid);

  // --- neighbor LSTM (dominant cost) ---
  for (int t = 0; t < kT; ++t) {
    launch_gemm(stream, Xnbr + (size_t)t * kNNB * kEMB, WihH, gates, nullptr, bsum1,
                kNNB, kG4, kEMB);
    launch_gemm(stream, hNbrH, WhhH, gates, gates, nullptr, kNNB, kG4, kENC);
    lstm_cell_k<<<g1(kNNB * kENC), 256, 0, stream>>>(gates, hNbr, cNbr, hNbrH,
                                                     noH + (size_t)t * kENC,
                                                     (long)kT * kENC, kNNB);
  }

  // --- neighbor attention pooling (soft_nbrs output) ---
  nbr_w_k<<<g1(kNNB * kT), 256, 0, stream>>>(noH, Wpre4, bpre4, wbuf);
  softmax_rows_k<<<g1(kNNB), 256, 0, stream>>>(wbuf, aNbr, out + kOUT * kB * 2, kNNB, kT);
  pool_nbr_k<<<g1(kNNB * kENC), 256, 0, stream>>>(noH, aNbr, nbrEnc);

  // --- masked scatter into social grid + second attention (soft_ha output) ---
  scan_mask_k<<<1, 1, 0, stream>>>(masks, idx);
  scatter_soc_k<<<g1(kB * kGH * kGW * kENC), 256, 0, stream>>>(idx, nbrEnc, newHs);
  set_hidden_col_k<<<g1(kB * kENC), 256, 0, stream>>>(nhid, newHs);
  w2_k<<<g1(kB * 40), 256, 0, stream>>>(newHs, Wpre4, bpre4, w2buf);
  softmax_rows_k<<<g1(kB), 256, 0, stream>>>(w2buf, aHa,
                                             out + kOUT * kB * 2 + kNNB * kT, kB, 40);
  pool_ha_k<<<g1(kB * kENC), 256, 0, stream>>>(newHs, aHa, encH);

  // --- decoder LSTM (constant input per step) + output projection ---
  launch_gemm(stream, encH, WihdH, xwd, nullptr, bsumd, kB, kG4, kENC);
  for (int t = 0; t < kOUT; ++t) {
    launch_gemm(stream, hDecH, WhhdH, gates, xwd, nullptr, kB, kG4, kDEC);
    lstm_cell_k<<<g1(kB * kDEC), 256, 0, stream>>>(gates, hDec, cDec, hDecH,
                                                   nullptr, 0, kB);
    dec_out_k<<<g1(kB * 2), 256, 0, stream>>>(hDec, Wop, bop, out + (size_t)t * kB * 2);
  }
}
